// MultimodalFusion_32117765439704
// MI455X (gfx1250) — matrix operand order repaired and checked by reading, not compiled, not run on hardware
//
#include <hip/hip_runtime.h>
#include <hip/hip_bf16.h>
#include <math.h>

// ---------------------------------------------------------------------------
// MultimodalFusion on MI455X (gfx1250): fused attention+MLP+LN, bf16 WMMA,
// TDM (tensor_load_to_lds) input staging.
// ---------------------------------------------------------------------------

typedef __bf16 bf16_t;
typedef bf16_t v16bf __attribute__((ext_vector_type(16)));
typedef bf16_t v8bf  __attribute__((ext_vector_type(8)));
typedef float  v8f   __attribute__((ext_vector_type(8)));
typedef unsigned int v4u __attribute__((ext_vector_type(4)));
typedef int    v8i   __attribute__((ext_vector_type(8)));
typedef int    v4i   __attribute__((ext_vector_type(4)));

#define NROWS   65536
#define TILE_M  16
#define NWG     (NROWS / TILE_M)   // 4096

// Build a 16-element bf16 WMMA fragment from two 16-byte loads.
__device__ __forceinline__ v16bf make_frag(const bf16_t* p0, const bf16_t* p1) {
    v8bf lo = *(const v8bf*)p0;
    v8bf hi = *(const v8bf*)p1;
    return __builtin_shufflevector(lo, hi, 0,1,2,3,4,5,6,7,8,9,10,11,12,13,14,15);
}

// A-matrix fragment (16x32, row-major, rowp = &A[m][0]); lane M = lane%16,
// elems 0..7 -> K = kb + 8*half + 0..7, elems 8..15 -> +16.  (ISA 7.12.2)
__device__ __forceinline__ v16bf frag_a(const bf16_t* rowp, int kb, int hl) {
    const bf16_t* p = rowp + kb + 8 * hl;
    return make_frag(p, p + 16);
}

// B-matrix fragment (32x16): B[k][n] = W[n][k], W row-major (N rows, K cols);
// lane N = lane%16, elems 0..7 -> K = kb + 8*half + 0..7, elems 8..15 -> +16.
__device__ __forceinline__ v16bf frag_b(const bf16_t* wrowp, int kb, int hl) {
    const bf16_t* p = wrowp + kb + 8 * hl;
    return make_frag(p, p + 16);
}

__device__ __forceinline__ v8f wmma_bf16(v16bf a, v16bf b, v8f c) {
    return __builtin_amdgcn_wmma_f32_16x16x32_bf16(false, a, false, b, (short)0, c, false, false);
}

__device__ __forceinline__ v8f zero_v8f() {
    v8f z = {0.f, 0.f, 0.f, 0.f, 0.f, 0.f, 0.f, 0.f};
    return z;
}

// TDM: DMA a 2D f32 tile (tile_d1 rows x tile_d0 elems, row stride stride0)
// from global memory into LDS at byte offset lds_off.  D# layout per
// CDNA5 ISA (count=1, type=2, data_size=4B).
// This toolchain exposes the 6-arg builtin form: groups 0..3 + extra + cpol.
__device__ __forceinline__ void tdm_load_2d_f32(const void* gaddr, unsigned lds_off,
                                                unsigned tensor_d0, unsigned tensor_d1,
                                                unsigned tile_d0, unsigned tile_d1,
                                                unsigned stride0) {
    unsigned long long ga = (unsigned long long)gaddr;
    v4u g0;
    g0.x = 1u;                                   // count=1, is_restore=0, gather=0
    g0.y = lds_off;                              // lds_addr (bytes)
    g0.z = (unsigned)(ga & 0xFFFFFFFFu);         // global_addr[31:0]
    g0.w = (unsigned)((ga >> 32) & 0x01FFFFFFu)  // global_addr[56:32]
         | (2u << 30);                           // type=2 ("image")
    v8i g1;
    g1[0] = (int)(2u << 16);                              // data_size=2 -> 4 bytes
    g1[1] = (int)((tensor_d0 & 0xFFFFu) << 16);           // tensor_dim0[15:0]
    g1[2] = (int)((tensor_d0 >> 16) | ((tensor_d1 & 0xFFFFu) << 16));
    g1[3] = (int)((tensor_d1 >> 16) | ((tile_d0 & 0xFFFFu) << 16));
    g1[4] = (int)(tile_d1 & 0xFFFFu);                     // tile_dim1; tile_dim2=0
    g1[5] = (int)stride0;                                 // tensor_dim0_stride[31:0]
    g1[6] = 0;                                            // stride0 hi, stride1 lo
    g1[7] = 0;
    v4i z4 = {0, 0, 0, 0};
    v8i z8 = {0, 0, 0, 0, 0, 0, 0, 0};
    __builtin_amdgcn_tensor_load_to_lds(g0, g1, z4, z4, z8, 0);
}

// f32 -> bf16 weight conversion (one-time, L2-resident afterwards)
__global__ void __launch_bounds__(256) cvt_f32_bf16(const float* __restrict__ src,
                                                    bf16_t* __restrict__ dst, int n) {
    int i = blockIdx.x * 256 + threadIdx.x;
    if (i < n) dst[i] = (bf16_t)src[i];
}

// ---------------------------------------------------------------------------
// Fused kernel: one workgroup handles 16 batch rows end-to-end.
// ---------------------------------------------------------------------------
__global__ void __launch_bounds__(256) fusion_kernel(
    const float* __restrict__ visual, const float* __restrict__ text,
    const bf16_t* __restrict__ Wqkv, const float* __restrict__ bqkv,
    const bf16_t* __restrict__ Wout, const float* __restrict__ bout,
    const bf16_t* __restrict__ W1,   const float* __restrict__ b1,
    const bf16_t* __restrict__ W2,   const float* __restrict__ b2,
    const float* __restrict__ gamma, const float* __restrict__ beta,
    float* __restrict__ out_fused,   float* __restrict__ out_weights)
{
    __shared__ __align__(16) char smem[65536];

    // Region A [0, 32768): X (2 x 16 x 512 bf16) during heads; flat (16x1024 bf16) after
    bf16_t* Xs   = (bf16_t*)smem;
    bf16_t* flat = (bf16_t*)smem;
    // Region B [32768, 65536):
    //   staging:      X f32 tile (16 x 512 = 32768B)          at +32768 (pre-head)
    //   during heads: qkv f32 (2x3x16x64 = 24576B)            at +32768
    //                 ctx bf16 (2x16x64 = 4096B)              at +57344
    //                 attn f32 (16x4 = 256B)                  at +61440
    //   after heads:  h bf16 (16x512 = 16384B)                at +32768
    //                 o f32  (16x256 = 16384B)                at +49152
    float*  stage   = (float*)(smem + 32768);
    float*  qkvf    = (float*)(smem + 32768);
    bf16_t* ctxb    = (bf16_t*)(smem + 57344);
    float*  attnLds = (float*)(smem + 61440);
    bf16_t* hbuf    = (bf16_t*)(smem + 32768);
    float*  obuf    = (float*)(smem + 49152);

    const int tid  = threadIdx.x;
    const int lane = tid & 31;
    const int wave = tid >> 5;
    const int ln   = lane & 15;   // fragment N / M index
    const int hl   = lane >> 4;   // fragment lane-half
    const int rowBase = blockIdx.x * TILE_M;

    // ---------------- Phase 0: TDM-stage X tiles, convert f32 -> bf16 ------
    for (int s = 0; s < 2; ++s) {
        const float* src = s ? text : visual;
        if (wave == 0) {
            tdm_load_2d_f32(src + (size_t)rowBase * 512, 32768u,
                            /*tensor_d0=*/512u, /*tensor_d1=*/16u,
                            /*tile_d0=*/512u, /*tile_d1=*/16u, /*stride0=*/512u);
            __builtin_amdgcn_s_wait_tensorcnt(0);
        }
        __syncthreads();   // DMA data visible to all waves
        #pragma unroll
        for (int j = 0; j < 8; ++j) {
            int v = tid + j * 256;                 // float4 index, 0..2047
            float4 f = ((const float4*)stage)[v];
            bf16_t* dst = Xs + s * 8192 + v * 4;
            dst[0] = (bf16_t)f.x; dst[1] = (bf16_t)f.y;
            dst[2] = (bf16_t)f.z; dst[3] = (bf16_t)f.w;
        }
        __syncthreads();   // staging free before next TDM / head loop
    }

    // Persistent attended accumulators: wave -> (seq sw, N-block nb of 128 cols)
    const int sw = wave >> 2;
    const int nb = wave & 3;
    v8f acc_att[8];
    #pragma unroll
    for (int j = 0; j < 8; ++j) acc_att[j] = zero_v8f();

    float wsum = 0.f;  // attn-weight mean accumulator (threads 0..31 only)

    // ---------------- Head loop -------------------------------------------
    for (int h = 0; h < 8; ++h) {
        // --- GEMM1: q/k/v. waves 0-3 -> seq0, waves 4-7 -> seq1;
        //     each wave: 3 (mat,nt) tiles sharing one A-frag per k-step ----
        {
            const int g = wave & 3;               // tile group within seq
            const bf16_t* Arow = Xs + sw * 8192 + ln * 512;
            const bf16_t* Brow[3];
            int   col[3];
            #pragma unroll
            for (int t3 = 0; t3 < 3; ++t3) {
                int t   = g * 3 + t3;             // 0..11
                int mat = t >> 2;                 // 0=q 1=k 2=v
                int nt  = t & 3;                  // 16-col tile within head dim
                col[t3] = mat * 512 + h * 64 + nt * 16 + ln;
                Brow[t3] = Wqkv + (size_t)col[t3] * 512;
            }
            v8f acc[3];
            #pragma unroll
            for (int t3 = 0; t3 < 3; ++t3) acc[t3] = zero_v8f();
            #pragma unroll 4
            for (int ks = 0; ks < 16; ++ks) {
                int kb = ks * 32;
                v16bf af = frag_a(Arow, kb, hl);  // shared across 3 tiles
                #pragma unroll
                for (int t3 = 0; t3 < 3; ++t3)
                    acc[t3] = wmma_bf16(af, frag_b(Brow[t3], kb, hl), acc[t3]);
            }
            #pragma unroll
            for (int t3 = 0; t3 < 3; ++t3) {
                int t   = g * 3 + t3;
                int mat = t >> 2;
                int nt  = t & 3;
                float bias = bqkv[col[t3]];
                #pragma unroll
                for (int i = 0; i < 8; ++i) {
                    int m = i + 8 * hl;
                    qkvf[((sw * 3 + mat) * 16 + m) * 64 + nt * 16 + ln] = acc[t3][i] + bias;
                }
            }
        }
        __syncthreads();

        // --- attention scores (seq len 2 -> softmax == sigmoid of diff) ---
        if (tid < 32) {
            int r = tid & 15, qi = tid >> 4;
            const float* q  = qkvf + ((qi * 3 + 0) * 16 + r) * 64;
            const float* k0 = qkvf + ((0 * 3 + 1) * 16 + r) * 64;
            const float* k1 = qkvf + ((1 * 3 + 1) * 16 + r) * 64;
            float s0 = 0.f, s1 = 0.f;
            #pragma unroll 8
            for (int d = 0; d < 64; ++d) { s0 += q[d] * k0[d]; s1 += q[d] * k1[d]; }
            float a0 = 1.f / (1.f + __expf((s1 - s0) * 0.125f));
            attnLds[r * 4 + qi * 2 + 0] = a0;
            attnLds[r * 4 + qi * 2 + 1] = 1.f - a0;
            wsum += a0;
        }
        __syncthreads();

        // --- ctx = attn @ v  (2 x 16 x 64, all 256 threads, 8 elems each) --
        {
            int r = tid & 15, c = tid >> 4;
            #pragma unroll
            for (int qi = 0; qi < 2; ++qi) {
                float a0 = attnLds[r * 4 + qi * 2 + 0];
                float a1 = attnLds[r * 4 + qi * 2 + 1];
                #pragma unroll
                for (int dd = 0; dd < 4; ++dd) {
                    int d = c * 4 + dd;
                    float v0 = qkvf[((0 * 3 + 2) * 16 + r) * 64 + d];
                    float v1 = qkvf[((1 * 3 + 2) * 16 + r) * 64 + d];
                    ctxb[qi * 1024 + r * 64 + d] = (bf16_t)(a0 * v0 + a1 * v1);
                }
            }
        }
        __syncthreads();

        // --- out_proj partial accumulate over this head's 64 K-columns ----
        {
            const bf16_t* Arow = ctxb + sw * 1024 + ln * 64;
            v16bf af[2];
            #pragma unroll
            for (int ks = 0; ks < 2; ++ks) af[ks] = frag_a(Arow, ks * 32, hl);
            #pragma unroll
            for (int j = 0; j < 8; ++j) {
                int n = nb * 128 + j * 16 + ln;
                const bf16_t* Brow = Wout + (size_t)n * 512 + h * 64;
                #pragma unroll
                for (int ks = 0; ks < 2; ++ks)
                    acc_att[j] = wmma_bf16(af[ks], frag_b(Brow, ks * 32, hl), acc_att[j]);
            }
        }
        __syncthreads();  // qkvf/ctx reused next head
    }

    // ---------------- attention weights output ----------------------------
    if (tid < 32) {
        int r = tid & 15, qi = tid >> 4;
        float m0 = wsum * 0.125f;  // mean over 8 heads
        out_weights[(size_t)(rowBase + r) * 4 + qi * 2 + 0] = m0;
        out_weights[(size_t)(rowBase + r) * 4 + qi * 2 + 1] = 1.f - m0;
    }

    // ---------------- attended + bias -> flat (bf16, aliases X) ------------
    #pragma unroll
    for (int j = 0; j < 8; ++j) {
        int n = nb * 128 + j * 16 + ln;
        float bias = bout[n];
        #pragma unroll
        for (int i = 0; i < 8; ++i) {
            int m = i + 8 * hl;
            flat[m * 1024 + sw * 512 + n] = (bf16_t)(acc_att[j][i] + bias);
        }
    }
    __syncthreads();

    // ---------------- MLP1: (16x1024)@(1024x512)^T + GELU ------------------
    {
        v8f acc[4];
        #pragma unroll
        for (int t = 0; t < 4; ++t) acc[t] = zero_v8f();
        const bf16_t* Arow = flat + ln * 1024;
        for (int ks = 0; ks < 32; ++ks) {
            int kb = ks * 32;
            v16bf af = frag_a(Arow, kb, hl);
            #pragma unroll
            for (int t = 0; t < 4; ++t) {
                int n = (wave * 4 + t) * 16 + ln;
                acc[t] = wmma_bf16(af, frag_b(W1 + (size_t)n * 1024, kb, hl), acc[t]);
            }
        }
        #pragma unroll
        for (int t = 0; t < 4; ++t) {
            int n = (wave * 4 + t) * 16 + ln;
            float bias = b1[n];
            #pragma unroll
            for (int i = 0; i < 8; ++i) {
                int m = i + 8 * hl;
                float v = acc[t][i] + bias;
                v = 0.5f * v * (1.f + erff(v * 0.70710678118f));  // exact GELU
                hbuf[m * 512 + n] = (bf16_t)v;
            }
        }
    }
    __syncthreads();

    // ---------------- MLP2: (16x512)@(512x256)^T ---------------------------
    {
        v8f acc[2];
        acc[0] = zero_v8f(); acc[1] = zero_v8f();
        const bf16_t* Arow = hbuf + ln * 512;
        for (int ks = 0; ks < 16; ++ks) {
            int kb = ks * 32;
            v16bf af = frag_a(Arow, kb, hl);
            #pragma unroll
            for (int t = 0; t < 2; ++t) {
                int n = (wave * 2 + t) * 16 + ln;
                acc[t] = wmma_bf16(af, frag_b(W2 + (size_t)n * 512, kb, hl), acc[t]);
            }
        }
        #pragma unroll
        for (int t = 0; t < 2; ++t) {
            int n = (wave * 2 + t) * 16 + ln;
            float bias = b2[n];
            #pragma unroll
            for (int i = 0; i < 8; ++i) {
                int m = i + 8 * hl;
                obuf[m * 256 + n] = acc[t][i] + bias;
            }
        }
    }
    __syncthreads();

    // ---------------- LayerNorm (wave handles 2 rows) ----------------------
    #pragma unroll
    for (int rr = 0; rr < 2; ++rr) {
        int r = wave * 2 + rr;
        const float* orow = obuf + r * 256;
        float s = 0.f, sq = 0.f;
        #pragma unroll
        for (int j = 0; j < 8; ++j) {
            float v = orow[lane + 32 * j];
            s += v; sq += v * v;
        }
        #pragma unroll
        for (int off = 16; off > 0; off >>= 1) {
            s  += __shfl_xor(s,  off, 32);
            sq += __shfl_xor(sq, off, 32);
        }
        float mean = s * (1.f / 256.f);
        float var  = sq * (1.f / 256.f) - mean * mean;
        float rstd = rsqrtf(var + 1e-5f);
        float* dst = out_fused + (size_t)(rowBase + r) * 256;
        #pragma unroll
        for (int j = 0; j < 8; ++j) {
            int c = lane + 32 * j;
            dst[c] = (orow[c] - mean) * rstd * gamma[c] + beta[c];
        }
    }
}

// ---------------------------------------------------------------------------
extern "C" void kernel_launch(void* const* d_in, const int* in_sizes, int n_in,
                              void* d_out, int out_size, void* d_ws, size_t ws_size,
                              hipStream_t stream) {
    const float* visual    = (const float*)d_in[0];
    const float* text      = (const float*)d_in[1];
    const float* in_proj_w = (const float*)d_in[2];   // (1536, 512)
    const float* in_proj_b = (const float*)d_in[3];
    const float* out_proj_w= (const float*)d_in[4];   // (512, 512)
    const float* out_proj_b= (const float*)d_in[5];
    const float* mlp_w1    = (const float*)d_in[6];   // (512, 1024)
    const float* mlp_b1    = (const float*)d_in[7];
    const float* mlp_w2    = (const float*)d_in[8];   // (256, 512)
    const float* mlp_b2    = (const float*)d_in[9];
    const float* ln_gamma  = (const float*)d_in[10];
    const float* ln_beta   = (const float*)d_in[11];

    // bf16 weight mirrors in workspace (~3.4 MB)
    bf16_t* Wqkv = (bf16_t*)d_ws;
    bf16_t* Wout = Wqkv + 786432;
    bf16_t* W1   = Wout + 262144;
    bf16_t* W2   = W1   + 524288;

    cvt_f32_bf16<<<(786432 + 255) / 256, 256, 0, stream>>>(in_proj_w,  Wqkv, 786432);
    cvt_f32_bf16<<<(262144 + 255) / 256, 256, 0, stream>>>(out_proj_w, Wout, 262144);
    cvt_f32_bf16<<<(524288 + 255) / 256, 256, 0, stream>>>(mlp_w1,     W1,   524288);
    cvt_f32_bf16<<<(131072 + 255) / 256, 256, 0, stream>>>(mlp_w2,     W2,   131072);

    float* out_fused   = (float*)d_out;
    float* out_weights = out_fused + (size_t)NROWS * 256;

    fusion_kernel<<<NWG, 256, 0, stream>>>(
        visual, text, Wqkv, in_proj_b, Wout, out_proj_b,
        W1, mlp_b1, W2, mlp_b2, ln_gamma, ln_beta,
        out_fused, out_weights);
}
